// HexTransformer_18648747999879
// MI455X (gfx1250) — matrix operand order repaired and checked by reading, not compiled, not run on hardware
//
#include <hip/hip_runtime.h>
#include <hip/hip_bf16.h>

// ---------------------------------------------------------------------------
// Hex-grid transformer forward, MI455X (gfx1250), wave32 + WMMA f16.
// K/V computed once per token (not per neighbor) -> 5x less in-proj work.
// GEMMs: v_wmma_f32_16x16x32_f16, f32 accumulate. Weight panels double-
// buffered in LDS via global_load_async_to_lds_b128; B-fragment reads and
// A-fragment global loads are software-pipelined against the WMMA stream.
// ---------------------------------------------------------------------------

typedef __attribute__((ext_vector_type(16))) _Float16 v16h;
typedef __attribute__((ext_vector_type(8)))  _Float16 v8h;
typedef __attribute__((ext_vector_type(8)))  float    v8f;
typedef __attribute__((ext_vector_type(4)))  int      v4i;

union AFrag { v16h v; v8h h[2]; };

__device__ __forceinline__ void async_copy_b128(const _Float16* g, _Float16* l) {
#if defined(__gfx1250__) && __has_builtin(__builtin_amdgcn_global_load_async_to_lds_b128)
  __builtin_amdgcn_global_load_async_to_lds_b128(
      (__attribute__((address_space(1))) v4i*)g,
      (__attribute__((address_space(3))) v4i*)l, 0, 0);
#else
  *(v8h*)l = *(const v8h*)g;
#endif
}

__device__ __forceinline__ void wait_async_lds() {
#if defined(__gfx1250__) && __has_builtin(__builtin_amdgcn_s_wait_asynccnt)
  __builtin_amdgcn_s_wait_asynccnt(0);
#elif defined(__gfx1250__)
  asm volatile("s_wait_asynccnt 0" ::: "memory");
#endif
}

// ---------------------------------------------------------------------------
// Weight pack: f32 (Cout x Cin, row-major; one matrix per blockIdx.y layer)
//  -> f16 WMMA B-fragment stream:
//  out[((nt*KT + kt)*32 + lane)*16 + j] = W[nt*16 + lane%16][kt*32 + (lane/16)*16 + j]
// ---------------------------------------------------------------------------
__global__ void __launch_bounds__(256) pack_weight_kernel(
    const float* __restrict__ W, _Float16* __restrict__ Out, int Cout, int Cin) {
  const size_t lay = (size_t)blockIdx.y * Cout * Cin;
  const int total = Cout * Cin;
  const int KT = Cin >> 5;
  for (int p = blockIdx.x * blockDim.x + threadIdx.x; p < total;
       p += gridDim.x * blockDim.x) {
    int j = p & 15;
    int lane = (p >> 4) & 31;
    int rest = p >> 9;
    int kt = rest % KT;
    int nt = rest / KT;
    int row = (nt << 4) + (lane & 15);
    int col = (kt << 5) + ((lane >> 4) << 4) + j;
    Out[lay + p] = (_Float16)W[lay + (size_t)row * Cin + col];
  }
}

// ---------------------------------------------------------------------------
// GEMM: Out[N x Cout] = act(A[N x K] * W^T + bias), f16 in/out, f32 accum.
// Block: 256 thr = 8 waves, tile 128 rows x (NT*16) cols, K-loop step 32.
// Weight panel (NT*512 halves) double-buffered in LDS (async-to-LDS).
// ---------------------------------------------------------------------------
template <int KT, int NT, bool RELU>
__global__ void __launch_bounds__(256) gemm_wmma_kernel(
    const _Float16* __restrict__ A, const _Float16* __restrict__ Wp,
    const float* __restrict__ bias, _Float16* __restrict__ Out, int Cout) {
  constexpr int PANEL = NT * 512;          // halves per K-step panel
  __shared__ _Float16 lb[2 * PANEL];
  const int K = KT * 32;
  const int lane = threadIdx.x & 31;
  const int wave = threadIdx.x >> 5;
  const int rowbase = blockIdx.x * 128 + wave * 16;
  const int gnt = blockIdx.y * NT;

  const v8f vzero = {0.f, 0.f, 0.f, 0.f, 0.f, 0.f, 0.f, 0.f};
  v8f acc[NT];
#pragma unroll
  for (int i = 0; i < NT; ++i) acc[i] = vzero;

  // A-fragment base for this lane: row = rowbase + lane%16,
  // lanes 0-15 read K {k..k+15}, lanes 16-31 K {k+16..k+31}
  const int r = rowbase + (lane & 15);
  const _Float16* arow = A + (size_t)r * K + ((lane >> 4) << 4);

  // staging slice for this thread: NT*2 halves (NT=16: 64B = 4x b128)
  const int base_half = threadIdx.x * (NT * 2);
  const int s_nt = base_half >> 9;
  const int s_off = base_half & 511;
  _Float16* lsd = lb + (s_nt << 9) + s_off;
  const _Float16* gsrc0 = Wp + (((size_t)(gnt + s_nt) * KT) << 9) + s_off;

  auto stage = [&](int kt, int buf) {
    const _Float16* gsrc = gsrc0 + ((size_t)kt << 9);
    _Float16* dst = lsd + buf * PANEL;
#pragma unroll
    for (int q = 0; q < (NT * 2) / 8; ++q)
      async_copy_b128(gsrc + q * 8, dst + q * 8);
  };

  // prologue: stage panel 0, prefetch A fragment 0
  stage(0, 0);
  AFrag af;
  af.h[0] = *(const v8h*)arow;
  af.h[1] = *(const v8h*)(arow + 8);

  for (int kt = 0; kt < KT; ++kt) {
    const int cur = kt & 1;
    wait_async_lds();        // own async writes (panel kt) complete
    __syncthreads();         // all waves' writes landed; prev-panel reads done

    if (kt + 1 < KT) stage(kt + 1, cur ^ 1);  // overlap next weight fetch

    // prefetch A fragment for kt+1 (global loads in flight during WMMAs)
    AFrag afn;
    if (kt + 1 < KT) {
      const _Float16* ap = arow + ((kt + 1) << 5);
      afn.h[0] = *(const v8h*)ap;
      afn.h[1] = *(const v8h*)(ap + 8);
    }

    // B-fragment 2-deep software pipeline: load nt+1 while wmma consumes nt
    const _Float16* lbase = lb + cur * PANEL + (lane << 4);
    AFrag bfr[2];
    bfr[0].h[0] = *(const v8h*)lbase;
    bfr[0].h[1] = *(const v8h*)(lbase + 8);
#pragma unroll
    for (int nt = 0; nt < NT; ++nt) {
      if (nt + 1 < NT) {
        const _Float16* lp = lbase + ((nt + 1) << 9);
        bfr[(nt + 1) & 1].h[0] = *(const v8h*)lp;
        bfr[(nt + 1) & 1].h[1] = *(const v8h*)(lp + 8);
      }
      acc[nt] = __builtin_amdgcn_wmma_f32_16x16x32_f16(
          false, af.v, false, bfr[nt & 1].v, (short)0, acc[nt], false, false);
    }
    if (kt + 1 < KT) af = afn;
  }

  // D layout: VGPR j holds (row = rowbase + j + 8*(lane>=16), col = base + lane%16)
  const int rw = rowbase + ((lane >> 4) << 3);
#pragma unroll
  for (int nt = 0; nt < NT; ++nt) {
    const int col = ((gnt + nt) << 4) + (lane & 15);
    const float bb = bias ? bias[col] : 0.f;
#pragma unroll
    for (int j = 0; j < 8; ++j) {
      float v = acc[nt][j] + bb;
      if (RELU) v = fmaxf(v, 0.f);
      Out[(size_t)(rw + j) * Cout + col] = (_Float16)v;
    }
  }
}

// ---------------------------------------------------------------------------
// Embed + sinusoidal position embedding -> x16 (f16)
// ---------------------------------------------------------------------------
__global__ void __launch_bounds__(256) embed_kernel(
    const int* __restrict__ grd, const float* __restrict__ gold,
    const float* __restrict__ emb, _Float16* __restrict__ x16) {
  const int t = blockIdx.x;
  const int c = threadIdx.x;
  const int b = t >> 12;
  const int hh = (t >> 6) & 63;
  const int ww = t & 63;
  float f;
  if (c < 254) f = emb[grd[t] * 254 + c];
  else         f = gold[(b << 1) + (c - 254)];
  const float kNeg = -9.210340371976184f / 128.f;  // -ln(10000)/half, half=128
  int d = (c < 128) ? c : (c - 128);
  float coord = (float)((c < 128) ? ww : hh);
  float pe;
  if (d < 64) pe = sinf(coord * expf((float)(d << 1) * kNeg));
  else        pe = cosf(coord * expf((float)((d - 64) << 1) * kNeg));
  x16[(size_t)t * 256 + c] = (_Float16)(f + pe);
}

// ---------------------------------------------------------------------------
// 7-neighbor hex attention. One thread per (token, head). OOB neighbors use
// pure-bias k/v (reference pads with zeros and does NOT mask them).
// ---------------------------------------------------------------------------
__global__ void __launch_bounds__(256) attn_kernel(
    const _Float16* __restrict__ qkv, const float* __restrict__ bvec,
    _Float16* __restrict__ o) {
  const int idx = blockIdx.x * 256 + threadIdx.x;
  const int t = idx >> 3;
  const int head = idx & 7;
  const int hh = (t >> 6) & 63;
  const int ww = t & 63;
  const int hb = head << 5;
  const _Float16* qp = qkv + (size_t)t * 768 + hb;
  float q[32];
#pragma unroll
  for (int d = 0; d < 32; ++d) q[d] = (float)qp[d];

  const int dqo[7] = {0, 1, -1, 0, 0, 1, -1};
  const int dro[7] = {0, 0, 0, 1, -1, -1, 1};
  int nbr[7]; bool ok[7]; float sc[7];
  const float scale = 0.1767766952966369f;  // 1/sqrt(32)
#pragma unroll
  for (int j = 0; j < 7; ++j) {
    int nh = hh + dro[j], nw = ww + dqo[j];
    ok[j] = ((unsigned)nh < 64u) && ((unsigned)nw < 64u);
    nbr[j] = (t & ~4095) | (nh << 6) | nw;
    float s = 0.f;
    if (ok[j]) {
      const _Float16* kp = qkv + (size_t)nbr[j] * 768 + 256 + hb;
#pragma unroll
      for (int d = 0; d < 32; ++d) s += q[d] * (float)kp[d];
    } else {
      const float* kb = bvec + 256 + hb;
#pragma unroll
      for (int d = 0; d < 32; ++d) s += q[d] * kb[d];
    }
    sc[j] = s * scale;
  }
  float m = sc[0];
#pragma unroll
  for (int j = 1; j < 7; ++j) m = fmaxf(m, sc[j]);
  float sum = 0.f;
#pragma unroll
  for (int j = 0; j < 7; ++j) { sc[j] = expf(sc[j] - m); sum += sc[j]; }
  const float inv = 1.f / sum;
  float oa[32];
#pragma unroll
  for (int d = 0; d < 32; ++d) oa[d] = 0.f;
#pragma unroll
  for (int j = 0; j < 7; ++j) {
    float w = sc[j] * inv;
    if (ok[j]) {
      const _Float16* vp = qkv + (size_t)nbr[j] * 768 + 512 + hb;
#pragma unroll
      for (int d = 0; d < 32; ++d) oa[d] += w * (float)vp[d];
    } else {
      const float* vb = bvec + 512 + hb;
#pragma unroll
      for (int d = 0; d < 32; ++d) oa[d] += w * vb[d];
    }
  }
  _Float16* op = o + (size_t)t * 256 + hb;
#pragma unroll
  for (int d = 0; d < 32; ++d) op[d] = (_Float16)oa[d];
}

// ---------------------------------------------------------------------------
// Residual + LayerNorm, one wave32 per token (8 tokens / block), f32 math.
// ---------------------------------------------------------------------------
__global__ void __launch_bounds__(256) ln_kernel(
    const _Float16* __restrict__ x, const _Float16* __restrict__ res,
    const float* __restrict__ g, const float* __restrict__ b,
    _Float16* __restrict__ out) {
  const int lane = threadIdx.x & 31;
  const int wave = threadIdx.x >> 5;
  const int t = blockIdx.x * 8 + wave;
  const _Float16* xp = x + (size_t)t * 256;
  float v[8];
  float s = 0.f, s2 = 0.f;
#pragma unroll
  for (int e = 0; e < 8; ++e) {
    int c = (e << 5) + lane;
    float f = (float)xp[c];
    if (res) f += (float)res[(size_t)t * 256 + c];
    v[e] = f; s += f; s2 += f * f;
  }
#pragma unroll
  for (int off = 16; off > 0; off >>= 1) {
    s  += __shfl_xor(s, off, 32);
    s2 += __shfl_xor(s2, off, 32);
  }
  const float mean = s * 0.00390625f;
  const float var  = s2 * 0.00390625f - mean * mean;
  const float rs   = rsqrtf(var + 1e-5f);
  _Float16* op = out + (size_t)t * 256;
#pragma unroll
  for (int e = 0; e < 8; ++e) {
    int c = (e << 5) + lane;
    op[c] = (_Float16)((v[e] - mean) * rs * g[c] + b[c]);
  }
}

// ---------------------------------------------------------------------------
// Mean-pool over H*W per (batch, channel)
// ---------------------------------------------------------------------------
__global__ void __launch_bounds__(256) pool_kernel(
    const _Float16* __restrict__ x, float* __restrict__ pooled) {
  const int b = blockIdx.x;
  const int c = threadIdx.x;
  float s = 0.f;
  const _Float16* xp = x + (((size_t)b << 12) << 8) + c;
  for (int p = 0; p < 4096; ++p) s += (float)xp[(size_t)p << 8];
  pooled[(b << 8) + c] = s * (1.f / 4096.f);
}

// ---------------------------------------------------------------------------
// Value head: relu(pooled @ cw1^T + cb1) @ cw2^T + cb2 -> out[3*65536 + b]
// ---------------------------------------------------------------------------
__global__ void __launch_bounds__(128) value_kernel(
    const float* __restrict__ pooled, const float* __restrict__ cw1,
    const float* __restrict__ cb1, const float* __restrict__ cw2,
    const float* __restrict__ cb2, float* __restrict__ outv) {
  __shared__ float red[128];
  const int j = threadIdx.x;
  for (int b = 0; b < 16; ++b) {
    float h = cb1[j];
    const float* pp = pooled + (b << 8);
    const float* wp = cw1 + (j << 8);
    for (int c = 0; c < 256; ++c) h += pp[c] * wp[c];
    h = fmaxf(h, 0.f);
    red[j] = h * cw2[j];
    __syncthreads();
    for (int off = 64; off > 0; off >>= 1) {
      if (j < off) red[j] += red[j + off];
      __syncthreads();
    }
    if (j == 0) outv[b] = red[0] + cb2[0];
    __syncthreads();
  }
}

// ---------------------------------------------------------------------------
// Policy tail: pol = h(512) @ pw2^T + pb2; scatter to (pol2, pol0, pol1)
// ---------------------------------------------------------------------------
__global__ void __launch_bounds__(256) polout_kernel(
    const _Float16* __restrict__ h, const float* __restrict__ pw2,
    const float* __restrict__ pb2, float* __restrict__ out) {
  const int t = blockIdx.x * 256 + threadIdx.x;
  const _Float16* hp = h + (size_t)t * 512;
  float a0 = pb2[0], a1 = pb2[1], a2 = pb2[2];
  for (int k = 0; k < 512; ++k) {
    float hv = (float)hp[k];
    a0 += hv * pw2[k];
    a1 += hv * pw2[512 + k];
    a2 += hv * pw2[1024 + k];
  }
  out[t]              = a2;   // pol[...,2]
  out[65536 + t]      = a0;   // pol[...,0]
  out[2 * 65536 + t]  = a1;   // pol[...,1]
}

// ---------------------------------------------------------------------------
// Host orchestration
// ---------------------------------------------------------------------------
extern "C" void kernel_launch(void* const* d_in, const int* in_sizes, int n_in,
                              void* d_out, int out_size, void* d_ws, size_t ws_size,
                              hipStream_t stream) {
  (void)in_sizes; (void)n_in; (void)out_size; (void)ws_size;
  constexpr int NTOK = 65536;   // 16*64*64
  constexpr int L = 8;

  const int*   grd  = (const int*)  d_in[0];
  const float* gold = (const float*)d_in[1];
  const float* emb  = (const float*)d_in[2];
  const float* in_w = (const float*)d_in[3];
  const float* in_b = (const float*)d_in[4];
  const float* outw = (const float*)d_in[5];
  const float* outb = (const float*)d_in[6];
  const float* l1w  = (const float*)d_in[7];
  const float* l1b  = (const float*)d_in[8];
  const float* l2w  = (const float*)d_in[9];
  const float* l2b  = (const float*)d_in[10];
  const float* n1g  = (const float*)d_in[11];
  const float* n1b  = (const float*)d_in[12];
  const float* n2g  = (const float*)d_in[13];
  const float* n2b  = (const float*)d_in[14];
  const float* fng  = (const float*)d_in[15];
  const float* fnb  = (const float*)d_in[16];
  const float* pw1  = (const float*)d_in[17];
  const float* pb1  = (const float*)d_in[18];
  const float* pw2  = (const float*)d_in[19];
  const float* pb2  = (const float*)d_in[20];
  const float* cw1  = (const float*)d_in[21];
  const float* cb1  = (const float*)d_in[22];
  const float* cw2  = (const float*)d_in[23];
  const float* cb2  = (const float*)d_in[24];
  float* out = (float*)d_out;

  char* ws = (char*)d_ws;
  auto alloc = [&](size_t bytes) {
    char* p = ws;
    ws += (bytes + 255) & ~(size_t)255;
    return p;
  };
  _Float16* x16    = (_Float16*)alloc((size_t)NTOK * 256 * 2);   // activation stream
  _Float16* bigbuf = (_Float16*)alloc((size_t)NTOK * 1024 * 2);  // qkv / ffn-hidden / pol-hidden
  _Float16* a16    = (_Float16*)alloc((size_t)NTOK * 256 * 2);   // branch output (pre-residual)
  _Float16* wqkv   = (_Float16*)alloc((size_t)L * 768 * 256 * 2);
  _Float16* wout   = (_Float16*)alloc((size_t)L * 256 * 256 * 2);
  _Float16* w1     = (_Float16*)alloc((size_t)L * 1024 * 256 * 2);
  _Float16* w2     = (_Float16*)alloc((size_t)L * 256 * 1024 * 2);
  _Float16* wp1    = (_Float16*)alloc((size_t)512 * 256 * 2);
  float*    pooled = (float*)   alloc((size_t)16 * 256 * 4);
  _Float16* o16 = bigbuf + (size_t)NTOK * 768;  // attn output lives past qkv

  // --- pack all weights to WMMA B-fragment f16 layout ---
  pack_weight_kernel<<<dim3(256, L), 256, 0, stream>>>(in_w, wqkv, 768, 256);
  pack_weight_kernel<<<dim3(256, L), 256, 0, stream>>>(outw, wout, 256, 256);
  pack_weight_kernel<<<dim3(256, L), 256, 0, stream>>>(l1w,  w1,  1024, 256);
  pack_weight_kernel<<<dim3(256, L), 256, 0, stream>>>(l2w,  w2,   256, 1024);
  pack_weight_kernel<<<dim3(256, 1), 256, 0, stream>>>(pw1,  wp1,  512, 256);

  // --- embedding + positional encoding ---
  embed_kernel<<<NTOK, 256, 0, stream>>>(grd, gold, emb, x16);

  for (int i = 0; i < L; ++i) {
    const _Float16* Wq = wqkv + (size_t)i * 768 * 256;
    const _Float16* Wo = wout + (size_t)i * 256 * 256;
    const _Float16* W1 = w1   + (size_t)i * 1024 * 256;
    const _Float16* W2 = w2   + (size_t)i * 256 * 1024;
    const float* bqkv = in_b + (size_t)i * 768;

    // QKV projection (once per token; neighbors reuse K/V)
    gemm_wmma_kernel<8, 16, false><<<dim3(512, 3), 256, 0, stream>>>(
        x16, Wq, bqkv, bigbuf, 768);
    // hex attention over 7 neighbors
    attn_kernel<<<2048, 256, 0, stream>>>(bigbuf, bqkv, o16);
    // output projection
    gemm_wmma_kernel<8, 16, false><<<dim3(512, 1), 256, 0, stream>>>(
        o16, Wo, outb + (size_t)i * 256, a16, 256);
    // residual + LN1
    ln_kernel<<<8192, 256, 0, stream>>>(x16, a16, n1g + i * 256, n1b + i * 256, x16);
    // FFN up (relu)
    gemm_wmma_kernel<8, 16, true><<<dim3(512, 4), 256, 0, stream>>>(
        x16, W1, l1b + (size_t)i * 1024, bigbuf, 1024);
    // FFN down
    gemm_wmma_kernel<32, 16, false><<<dim3(512, 1), 256, 0, stream>>>(
        bigbuf, W2, l2b + (size_t)i * 256, a16, 256);
    // residual + LN2
    ln_kernel<<<8192, 256, 0, stream>>>(x16, a16, n2g + i * 256, n2b + i * 256, x16);
  }

  // final LN
  ln_kernel<<<8192, 256, 0, stream>>>(x16, nullptr, fng, fnb, x16);
  // value head
  pool_kernel<<<16, 256, 0, stream>>>(x16, pooled);
  value_kernel<<<1, 128, 0, stream>>>(pooled, cw1, cb1, cw2, cb2, out + 3 * 65536);
  // policy head
  gemm_wmma_kernel<8, 16, true><<<dim3(512, 2), 256, 0, stream>>>(
      x16, wp1, pb1, bigbuf, 512);
  polout_kernel<<<256, 256, 0, stream>>>(bigbuf, pw2, pb2, out);
}
